// SparseMoEEncoder_12249246729056
// MI455X (gfx1250) — matrix operand order repaired and checked by reading, not compiled, not run on hardware
//
#include <hip/hip_runtime.h>
#include <cstddef>

// ---------------------------------------------------------------------------
// SparseMoE patch encoder for gfx1250 (MI455X, wave32, WMMA).
// Heavy GEMMs (gate-1, pw2, pw3) use v_wmma_f32_16x16x32_bf16.
// ---------------------------------------------------------------------------

typedef __bf16 bf16;
typedef __attribute__((ext_vector_type(8)))  __bf16 v8bf;
typedef __attribute__((ext_vector_type(16))) __bf16 v16bf;
typedef __attribute__((ext_vector_type(8)))  float  v8f;

#define BN_  8192      // B * patches (32 * 256)
#define GIN_ 768
#define GH_  128
#define E_   8
#define D_   256

// ---- bf16 helpers (bit-exact RNE, no reliance on __bf16 arithmetic) -------
static __device__ __forceinline__ bf16 f2bf(float f) {
  union { float f; unsigned u; } x; x.f = f;
  unsigned r = x.u + 0x7FFFu + ((x.u >> 16) & 1u);
  union { unsigned short s; bf16 b; } y; y.s = (unsigned short)(r >> 16);
  return y.b;
}
static __device__ __forceinline__ float bf2f(bf16 b) {
  union { unsigned short s; bf16 b; } y; y.b = b;
  union { unsigned u; float f; } x; x.u = ((unsigned)y.s) << 16;
  return x.f;
}

// ---- WMMA fragment loaders (16-bit 16x32 A, 32x16 B; wave32) --------------
// A row-major [M][K]; lane holds row m=lane&15, half=lane>>4.
static __device__ __forceinline__ v16bf load_a_frag(const bf16* A, int row, int K,
                                                    int ks, int half) {
  return *reinterpret_cast<const v16bf*>(A + (size_t)row * K + ks + half * 16);
}
// B[k][n] = Wt[n][k] with Wt row-major [N][K]; b[i] = Wt[n][ks + half*16 + i].
static __device__ __forceinline__ v16bf load_b_frag(const bf16* Wt, int col, int K,
                                                    int ks, int half) {
  return *reinterpret_cast<const v16bf*>(Wt + (size_t)col * K + ks + half * 16);
}

// ---------------------------------------------------------------------------
// Kernel 1: unfold patches -> bf16 gate A-matrix [BN, 768], col = (c,pi,pj)
// ---------------------------------------------------------------------------
__global__ void prep_patches_kernel(const float* __restrict__ x, bf16* __restrict__ A) {
  int t = blockIdx.x * 256 + threadIdx.x;        // BN*768 threads
  int n = t / GIN_, col = t % GIN_;
  int c = col >> 8, r = col & 255, pi = r >> 4, pj = r & 15;
  int b = n >> 8, pp = n & 255, hp = pp >> 4, wp = pp & 15;
  float v = x[(((size_t)b * 3 + c) * 256 + (hp * 16 + pi)) * 256 + (wp * 16 + pj)];
  A[t] = f2bf(v);
}

__global__ void zero_accum_kernel(float* __restrict__ acc) {
  acc[blockIdx.x * 256 + threadIdx.x] = 0.0f;    // BN*256 threads
}

// fp32 -> bf16 weight conversion: g_w1 (98304), pw2_w (65536), pw3_w (262144)
__global__ void convert_weights_kernel(const float* __restrict__ g_w1,
                                       const float* __restrict__ pw2w,
                                       const float* __restrict__ pw3w,
                                       bf16* __restrict__ w1bf,
                                       bf16* __restrict__ pw2bf,
                                       bf16* __restrict__ pw3bf) {
  int t = blockIdx.x * 256 + threadIdx.x;        // 425984 threads exactly
  if (t < 98304) {
    w1bf[t] = f2bf(g_w1[t]);
  } else if (t < 98304 + 65536) {
    int i = t - 98304; pw2bf[i] = f2bf(pw2w[i]);
  } else {
    int i = t - 163840; pw3bf[i] = f2bf(pw3w[i]);
  }
}

// ---------------------------------------------------------------------------
// WMMA GEMM, N = 128: Out = relu(A[M,K] * Wt[128,K]^T + bias), bf16 out.
// grid.x = M/16; 8 waves per block, wave w owns columns [16w, 16w+16).
// Used for gate layer 1 (K=768) and pw2 (K=64).
// ---------------------------------------------------------------------------
__global__ void gemm_relu_n128_kernel(const bf16* __restrict__ A,
                                      const bf16* __restrict__ Wt,
                                      const float* __restrict__ bias,
                                      bf16* __restrict__ Out, int K) {
  const int tid  = threadIdx.x;
  const int lane = tid & 31, wave = tid >> 5;
  const int half = lane >> 4, l15 = lane & 15;
  const int mtile = blockIdx.x;
  const int row = mtile * 16 + l15;
  const int col = wave * 16 + l15;
  v8f acc = {0, 0, 0, 0, 0, 0, 0, 0};
  for (int ks = 0; ks < K; ks += 32) {
    v16bf afrag = load_a_frag(A, row, K, ks, half);
    v16bf bfrag = load_b_frag(Wt, col, K, ks, half);
    acc = __builtin_amdgcn_wmma_f32_16x16x32_bf16(false, afrag, false, bfrag,
                                                  (short)0, acc, false, false);
  }
  float bv = bias[col];
#pragma unroll
  for (int v = 0; v < 8; ++v) {
    int m = mtile * 16 + v + 8 * half;           // C/D layout: row = v + 8*half
    float r = acc[v] + bv;
    Out[(size_t)m * 128 + col] = f2bf(r > 0.0f ? r : 0.0f);
  }
}

// ---------------------------------------------------------------------------
// Gate layer 2 + top-2 softmax routing. One thread per patch.
// ---------------------------------------------------------------------------
__global__ void gate_route_kernel(const bf16* __restrict__ Hb,
                                  const float* __restrict__ w2,
                                  const float* __restrict__ b2,
                                  float* __restrict__ routing) {
  int n = blockIdx.x * 256 + threadIdx.x;        // 8192 threads
  float logits[E_];
#pragma unroll
  for (int e = 0; e < E_; ++e) logits[e] = b2[e];
  const bf16* hrow = Hb + (size_t)n * GH_;
  for (int k = 0; k < GH_; ++k) {
    float hv = bf2f(hrow[k]);
#pragma unroll
    for (int e = 0; e < E_; ++e) logits[e] += hv * w2[e * GH_ + k];
  }
  int bi = 0; float bv = logits[0];
#pragma unroll
  for (int e = 1; e < E_; ++e) if (logits[e] > bv) { bv = logits[e]; bi = e; }
  int si = -1; float sv = -3.4e38f;
#pragma unroll
  for (int e = 0; e < E_; ++e) if (e != bi && logits[e] > sv) { sv = logits[e]; si = e; }
  float e2  = __expf(sv - bv);
  float inv = 1.0f / (1.0f + e2);
#pragma unroll
  for (int e = 0; e < E_; ++e) {
    float r = (e == bi) ? inv : ((e == si) ? e2 * inv : 0.0f);
    routing[n * E_ + e] = r;
  }
}

// ---------------------------------------------------------------------------
// Expert stage 1: depthwise 3x3 s2 (3 ch, 16x16 -> 8x8) + pw 3->64, fused.
// One thread per (patch, output position). a1: [n*64 + oy*8+ox][64ch] bf16.
// ---------------------------------------------------------------------------
__global__ void dw1_pw1_kernel(const float* __restrict__ x,
                               const float* __restrict__ dw1w, const float* __restrict__ dw1b,
                               const float* __restrict__ pw1w, const float* __restrict__ pw1b,
                               bf16* __restrict__ a1, int e) {
  int t = blockIdx.x * 256 + threadIdx.x;        // BN*64 threads
  int n = t >> 6, opos = t & 63;
  int oy = opos >> 3, ox = opos & 7;
  int b = n >> 8, pp = n & 255, hp = pp >> 4, wp = pp & 15;
  float s[3];
#pragma unroll
  for (int c = 0; c < 3; ++c) {
    float acc = dw1b[e * 3 + c];
#pragma unroll
    for (int ky = 0; ky < 3; ++ky) {
      int iy = oy * 2 - 1 + ky;
      if (iy < 0 || iy > 15) continue;
#pragma unroll
      for (int kx = 0; kx < 3; ++kx) {
        int ix = ox * 2 - 1 + kx;
        if (ix < 0 || ix > 15) continue;
        float xv = x[(((size_t)b * 3 + c) * 256 + (hp * 16 + iy)) * 256 + (wp * 16 + ix)];
        acc += xv * dw1w[(e * 3 + c) * 9 + ky * 3 + kx];
      }
    }
    s[c] = acc > 0.0f ? acc : 0.0f;
  }
  bf16* outp = a1 + ((size_t)n * 64 + opos) * 64;
  for (int oc = 0; oc < 64; ++oc) {
    const float* w = pw1w + (e * 64 + oc) * 3;
    float a = pw1b[e * 64 + oc] + s[0] * w[0] + s[1] * w[1] + s[2] * w[2];
    outp[oc] = f2bf(a > 0.0f ? a : 0.0f);
  }
}

// Depthwise 3x3 s2 on 8x8 -> 4x4, 64 ch. d2: [n*16 + oy*4+ox][64] bf16.
__global__ void dw2_kernel(const bf16* __restrict__ a1,
                           const float* __restrict__ dw2w, const float* __restrict__ dw2b,
                           bf16* __restrict__ d2, int e) {
  int t = blockIdx.x * 256 + threadIdx.x;        // BN*16*64 threads
  int ch = t & 63; int rest = t >> 6;
  int opos = rest & 15; int n = rest >> 4;
  int oy = opos >> 2, ox = opos & 3;
  float acc = dw2b[e * 64 + ch];
#pragma unroll
  for (int ky = 0; ky < 3; ++ky) {
    int iy = oy * 2 - 1 + ky;
    if (iy < 0 || iy > 7) continue;
#pragma unroll
    for (int kx = 0; kx < 3; ++kx) {
      int ix = ox * 2 - 1 + kx;
      if (ix < 0 || ix > 7) continue;
      acc += bf2f(a1[((size_t)n * 64 + iy * 8 + ix) * 64 + ch]) *
             dw2w[(e * 64 + ch) * 9 + ky * 3 + kx];
    }
  }
  d2[((size_t)n * 16 + opos) * 64 + ch] = f2bf(acc > 0.0f ? acc : 0.0f);
}

// Depthwise 3x3 s2 on 4x4 -> 2x2, 128 ch. d3: [n*4 + oy*2+ox][128] bf16.
__global__ void dw3_kernel(const bf16* __restrict__ a2,
                           const float* __restrict__ dw3w, const float* __restrict__ dw3b,
                           bf16* __restrict__ d3, int e) {
  int t = blockIdx.x * 256 + threadIdx.x;        // BN*4*128 threads
  int ch = t & 127; int rest = t >> 7;
  int opos = rest & 3; int n = rest >> 2;
  int oy = opos >> 1, ox = opos & 1;
  float acc = dw3b[e * 128 + ch];
#pragma unroll
  for (int ky = 0; ky < 3; ++ky) {
    int iy = oy * 2 - 1 + ky;
    if (iy < 0 || iy > 3) continue;
#pragma unroll
    for (int kx = 0; kx < 3; ++kx) {
      int ix = ox * 2 - 1 + kx;
      if (ix < 0 || ix > 3) continue;
      acc += bf2f(a2[((size_t)n * 16 + iy * 4 + ix) * 128 + ch]) *
             dw3w[(e * 128 + ch) * 9 + ky * 3 + kx];
    }
  }
  d3[((size_t)n * 4 + opos) * 128 + ch] = f2bf(acc > 0.0f ? acc : 0.0f);
}

// ---------------------------------------------------------------------------
// pw3 WMMA GEMM (M=BN*4, K=128, N=256) fused with relu, 2x2 avg-pool and
// routing-weighted accumulation. Block tile = 16 rows (= 4 patches); each
// wave handles two 16-col tiles. Rows v..v+3 of each patch live in one lane,
// so pooling is lane-local; each (patch, col) is owned by exactly one lane.
// ---------------------------------------------------------------------------
__global__ void pw3_pool_kernel(const bf16* __restrict__ A,
                                const bf16* __restrict__ Wt,
                                const float* __restrict__ bias,
                                const float* __restrict__ routing,
                                float* __restrict__ accum, int e) {
  const int tid  = threadIdx.x;
  const int lane = tid & 31, wave = tid >> 5;
  const int half = lane >> 4, l15 = lane & 15;
  const int mtile = blockIdx.x;                  // 2048 tiles
  const int row = mtile * 16 + l15;
  for (int nt = wave; nt < 16; nt += 8) {
    const int col = nt * 16 + l15;
    v8f acc = {0, 0, 0, 0, 0, 0, 0, 0};
    for (int ks = 0; ks < 128; ks += 32) {
      v16bf afrag = load_a_frag(A, row, 128, ks, half);
      v16bf bfrag = load_b_frag(Wt, col, 128, ks, half);
      acc = __builtin_amdgcn_wmma_f32_16x16x32_bf16(false, afrag, false, bfrag,
                                                    (short)0, acc, false, false);
    }
    float bv = bias[col];
    float r[8];
#pragma unroll
    for (int v = 0; v < 8; ++v) { float q = acc[v] + bv; r[v] = q > 0.0f ? q : 0.0f; }
    // rows m = v + 8*half; patch = m>>2, pos = m&3 -> lane-local pooling
    float p0 = (r[0] + r[1] + r[2] + r[3]) * 0.25f;   // patch half*2 + 0
    float p1 = (r[4] + r[5] + r[6] + r[7]) * 0.25f;   // patch half*2 + 1
    int gp0 = mtile * 4 + half * 2;
    int gp1 = gp0 + 1;
    float w0 = routing[gp0 * E_ + e];
    float w1 = routing[gp1 * E_ + e];
    float* o0 = accum + (size_t)gp0 * D_ + col;
    float* o1 = accum + (size_t)gp1 * D_ + col;
    *o0 += w0 * p0;                                   // expert kernels are
    *o1 += w1 * p1;                                   // stream-serialized
  }
}

// Final layout: features[b][d][hp][wp] = accum[n][d]; loss scalar = 0.
__global__ void output_kernel(const float* __restrict__ accum, float* __restrict__ out) {
  int t = blockIdx.x * 256 + threadIdx.x;        // 2,097,152 threads
  int wp = t & 15, hp = (t >> 4) & 15, d = (t >> 8) & 255, b = t >> 16;
  int n = (b << 8) + (hp << 4) + wp;
  out[t] = accum[(size_t)n * D_ + d];
  if (t == 0) out[2097152] = 0.0f;               // load_balance_loss (eval)
}

// ---------------------------------------------------------------------------
// Workspace layout (~150 MB), all offsets 256B-aligned.
// ---------------------------------------------------------------------------
static constexpr size_t OFF_GATEA = 0;                                        // bf16 [BN,768]
static constexpr size_t OFF_H     = OFF_GATEA + (size_t)BN_ * GIN_ * 2;       // bf16 [BN,128]
static constexpr size_t OFF_ROUT  = OFF_H     + (size_t)BN_ * GH_ * 2;        // f32  [BN,8]
static constexpr size_t OFF_W1BF  = OFF_ROUT  + (size_t)BN_ * E_ * 4;         // bf16 [128,768]
static constexpr size_t OFF_PW2BF = OFF_W1BF  + (size_t)GH_ * GIN_ * 2;       // bf16 [8,128,64]
static constexpr size_t OFF_PW3BF = OFF_PW2BF + (size_t)E_ * 128 * 64 * 2;    // bf16 [8,256,128]
static constexpr size_t OFF_A1    = OFF_PW3BF + (size_t)E_ * 256 * 128 * 2;   // bf16 [BN*64,64]
static constexpr size_t OFF_D2    = OFF_A1    + (size_t)BN_ * 64 * 64 * 2;    // bf16 [BN*16,64]
static constexpr size_t OFF_A2    = OFF_D2    + (size_t)BN_ * 16 * 64 * 2;    // bf16 [BN*16,128]
static constexpr size_t OFF_D3    = OFF_A2    + (size_t)BN_ * 16 * 128 * 2;   // bf16 [BN*4,128]
static constexpr size_t OFF_ACC   = OFF_D3    + (size_t)BN_ * 4 * 128 * 2;    // f32  [BN,256]

extern "C" void kernel_launch(void* const* d_in, const int* in_sizes, int n_in,
                              void* d_out, int out_size, void* d_ws, size_t ws_size,
                              hipStream_t stream) {
  (void)in_sizes; (void)n_in; (void)out_size; (void)ws_size;
  const float* x     = (const float*)d_in[0];
  const float* g_w1  = (const float*)d_in[1];
  const float* g_b1  = (const float*)d_in[2];
  const float* g_w2  = (const float*)d_in[3];
  const float* g_b2  = (const float*)d_in[4];
  const float* dw1_w = (const float*)d_in[5];
  const float* dw1_b = (const float*)d_in[6];
  const float* pw1_w = (const float*)d_in[7];
  const float* pw1_b = (const float*)d_in[8];
  const float* dw2_w = (const float*)d_in[9];
  const float* dw2_b = (const float*)d_in[10];
  const float* pw2_w = (const float*)d_in[11];
  const float* pw2_b = (const float*)d_in[12];
  const float* dw3_w = (const float*)d_in[13];
  const float* dw3_b = (const float*)d_in[14];
  const float* pw3_w = (const float*)d_in[15];
  const float* pw3_b = (const float*)d_in[16];
  float* out = (float*)d_out;
  char*  ws  = (char*)d_ws;

  bf16*  gateA = (bf16*)(ws + OFF_GATEA);
  bf16*  hbuf  = (bf16*)(ws + OFF_H);
  float* rout  = (float*)(ws + OFF_ROUT);
  bf16*  w1bf  = (bf16*)(ws + OFF_W1BF);
  bf16*  pw2bf = (bf16*)(ws + OFF_PW2BF);
  bf16*  pw3bf = (bf16*)(ws + OFF_PW3BF);
  bf16*  a1    = (bf16*)(ws + OFF_A1);
  bf16*  d2    = (bf16*)(ws + OFF_D2);
  bf16*  a2    = (bf16*)(ws + OFF_A2);
  bf16*  d3    = (bf16*)(ws + OFF_D3);
  float* accum = (float*)(ws + OFF_ACC);

  // Prep: patches, accumulator, bf16 weights.
  prep_patches_kernel<<<24576, 256, 0, stream>>>(x, gateA);      // BN*768
  zero_accum_kernel<<<8192, 256, 0, stream>>>(accum);            // BN*256
  convert_weights_kernel<<<1664, 256, 0, stream>>>(g_w1, pw2_w, pw3_w,
                                                   w1bf, pw2bf, pw3bf);

  // Gate: WMMA GEMM [8192,768]x[768,128] -> relu, then top-2 routing.
  gemm_relu_n128_kernel<<<512, 256, 0, stream>>>(gateA, w1bf, g_b1, hbuf, GIN_);
  gate_route_kernel<<<32, 256, 0, stream>>>(hbuf, g_w2, g_b2, rout);

  // Dense experts (reference runs all experts), workspace reused per expert.
  for (int e = 0; e < E_; ++e) {
    dw1_pw1_kernel<<<2048, 256, 0, stream>>>(x, dw1_w, dw1_b, pw1_w, pw1_b, a1, e);
    dw2_kernel<<<32768, 256, 0, stream>>>(a1, dw2_w, dw2_b, d2, e);
    gemm_relu_n128_kernel<<<8192, 256, 0, stream>>>(d2, pw2bf + (size_t)e * 128 * 64,
                                                    pw2_b + e * 128, a2, 64);
    dw3_kernel<<<16384, 256, 0, stream>>>(a2, dw3_w, dw3_b, d3, e);
    pw3_pool_kernel<<<2048, 256, 0, stream>>>(d3, pw3bf + (size_t)e * 256 * 128,
                                              pw3_b + e * 256, rout, accum, e);
  }

  output_kernel<<<8192, 256, 0, stream>>>(accum, out);
}
